// RelativeMultiHeadAttention_7937099563323
// MI455X (gfx1250) — matrix operand order repaired and checked by reading, not compiled, not run on hardware
//
#include <hip/hip_runtime.h>
#include <math.h>

// ---------------------------------------------------------------------------
// Transformer-XL relative multi-head attention for MI455X (gfx1250).
// All matmuls via v_wmma_f32_16x16x32_bf16 (wave32 WMMA, f32 accumulate).
// B=16, S=512, D=1024, H=16, DH=64.
// ---------------------------------------------------------------------------

#define BQ      16
#define SEQ     512
#define HEADS   16
#define DMODEL  1024
#define DHEAD   64
#define ROWS    (BQ * SEQ)          // 8192

typedef __bf16 bf16;
typedef __attribute__((ext_vector_type(8)))  __bf16 bf16x8;
typedef __attribute__((ext_vector_type(16))) __bf16 bf16x16;
typedef __attribute__((ext_vector_type(8)))  float  f32x8;
typedef int v4i __attribute__((vector_size(16)));   // matches builtin signature

union FragU { bf16x16 v; bf16x8 h[2]; };

// CDNA5 async global->LDS staging (guarded; falls back to plain b128 copies).
#if defined(__has_builtin)
#  if __has_builtin(__builtin_amdgcn_global_load_async_to_lds_b128) && \
      __has_builtin(__builtin_amdgcn_s_wait_asynccnt)
#    define USE_ASYNC_LDS 1
#  endif
#endif
#ifndef USE_ASYNC_LDS
#  define USE_ASYNC_LDS 0
#endif

__device__ __forceinline__ void stage16B(const bf16* src, bf16* dst) {
#if USE_ASYNC_LDS
  __builtin_amdgcn_global_load_async_to_lds_b128(
      (__attribute__((address_space(1))) v4i*)(src),
      (__attribute__((address_space(3))) v4i*)(dst),
      /*offset=*/0, /*cpol=*/0);
#else
  *(bf16x8*)dst = *(const bf16x8*)src;
#endif
}

__device__ __forceinline__ void stage_fence() {
#if USE_ASYNC_LDS
  __builtin_amdgcn_s_wait_asynccnt(0);
#endif
  __syncthreads();
}

__device__ __forceinline__ f32x8 wmma_bf16(bf16x16 a, bf16x16 b, f32x8 c) {
  // D = A(16x32 bf16) x B(32x16 bf16) + C(16x16 f32)
  return __builtin_amdgcn_wmma_f32_16x16x32_bf16(
      /*neg_a=*/false, a, /*neg_b=*/false, b,
      /*c_mod=*/(short)0, c, /*reuse_a=*/false, /*reuse_b=*/false);
}

// A-fragment (16x32, row-major source, leading dim ldk).
// Lane l: M = l%16, K = 8*(l/16) + (e%8) + 16*(e/8)  -> two contiguous 8-elem runs.
__device__ __forceinline__ bf16x16 load_a_rm(const bf16* base, int ldk, int lane) {
  int M  = lane & 15;
  int kb = (lane >> 4) << 3;            // 0 or 8
  const bf16* p = base + (size_t)M * ldk + kb;
  FragU f;
  f.h[0] = *(const bf16x8*)(p);
  f.h[1] = *(const bf16x8*)(p + 16);
  return f.v;
}

// B-fragment where memory holds the 16(n) x 32(k) tile row-major as [n][k]
// (i.e. B^T). Lane l: N = l%16, K = 8*(l/16) + (e%8) + 16*(e/8) -> two contiguous 8-elem runs.
__device__ __forceinline__ bf16x16 load_b_nt(const bf16* base, int ldk, int lane) {
  int N  = lane & 15;
  int ko = (lane >> 4) << 3;            // 0 or 8
  const bf16* p = base + (size_t)N * ldk + ko;
  FragU f;
  f.h[0] = *(const bf16x8*)(p);
  f.h[1] = *(const bf16x8*)(p + 16);
  return f.v;
}

// A-fragment from LDS row-major [16][stride] tile.
__device__ __forceinline__ bf16x16 load_a_lds(const bf16* base, int stride, int lane) {
  int M  = lane & 15;
  int kb = (lane >> 4) << 3;
  const bf16* p = base + M * stride + kb;
  FragU f;
  f.h[0] = *(const bf16x8*)(p);
  f.h[1] = *(const bf16x8*)(p + 16);
  return f.v;
}

// ---------------------------------------------------------------------------
// f32 -> bf16 conversion (plain, and transposing for weight matrices).
// ---------------------------------------------------------------------------
__global__ void cvt_kernel(const float* __restrict__ s, bf16* __restrict__ d, int n) {
  int t = blockIdx.x * blockDim.x + threadIdx.x;
  if (t < n) d[t] = (bf16)s[t];
}

// d[c][r] = s[r][c] : store W^T so GEMM B-tiles are K-contiguous in memory.
__global__ void cvt_t_kernel(const float* __restrict__ s, bf16* __restrict__ d,
                             int R, int C) {
  int t = blockIdx.x * blockDim.x + threadIdx.x;
  if (t >= R * C) return;
  int r = t / C, c = t - r * C;
  d[(size_t)c * R + r] = (bf16)s[t];
}

// ---------------------------------------------------------------------------
// Tiled WMMA GEMM: C[M,N] = A[M,K] * W[K,N], W given pre-transposed WT[N][K].
// 256 threads = 8 waves; 128x128 output tile per block (each wave 16x128).
// B panel (128n x 32k) staged K-contiguous into LDS via async-to-LDS.
// ---------------------------------------------------------------------------
template <bool OUT_F32>
__global__ __launch_bounds__(256)
void gemm_bf16(const bf16* __restrict__ A, const bf16* __restrict__ WT,
               void* __restrict__ Cout, int M, int N, int K) {
  __shared__ bf16 wt[128][40];   // wt[n_loc][k_loc], padded stride (16B-aligned rows)
  const int lane = threadIdx.x & 31;
  const int wave = threadIdx.x >> 5;
  const int n0 = blockIdx.x * 128;
  const int m0 = blockIdx.y * 128;
  const int i0 = m0 + wave * 16;

  f32x8 acc[8] = {};
  for (int k0 = 0; k0 < K; k0 += 32) {
    __syncthreads();
    // Stage WT[n0..n0+127][k0..k0+31]: 512 x 16-byte chunks, 2 per thread.
#pragma unroll
    for (int it = 0; it < 2; ++it) {
      int idx = threadIdx.x + it * 256;
      int nl = idx >> 2, kc = (idx & 3) << 3;
      stage16B(WT + (size_t)(n0 + nl) * K + k0 + kc, &wt[nl][kc]);
    }
    stage_fence();
    if (k0 + 32 < K)
      __builtin_prefetch(A + (size_t)(i0 + (lane & 15)) * K + k0 + 32, 0, 1);
    bf16x16 a = load_a_rm(A + (size_t)i0 * K + k0, K, lane);
#pragma unroll
    for (int n = 0; n < 8; ++n) {
      bf16x16 bv = load_b_nt(&wt[n * 16][0], 40, lane);
      acc[n] = wmma_bf16(a, bv, acc[n]);
    }
  }
  const int Nl = lane & 15;
  const int Mb = (lane >> 4) << 3;
#pragma unroll
  for (int n = 0; n < 8; ++n)
#pragma unroll
    for (int r = 0; r < 8; ++r) {
      size_t off = (size_t)(i0 + Mb + r) * N + n0 + n * 16 + Nl;
      if (OUT_F32) ((float*)Cout)[off] = acc[n][r];
      else         ((bf16*)Cout)[off]  = (bf16)acc[n][r];
    }
}

// ---------------------------------------------------------------------------
// Bias rows: EF[b,h,j] = u[h]·k[b,j,h,:]   GH[h,j] = v[h]·kr[j,h,:]
// (_relative_shift of the [1,h,1,j] GH tensor is the identity.)
// ---------------------------------------------------------------------------
__global__ void ef_kernel(const float* __restrict__ u, const bf16* __restrict__ Kb,
                          float* __restrict__ EFp) {
  int t = blockIdx.x * blockDim.x + threadIdx.x;
  if (t >= BQ * HEADS * SEQ) return;
  int j = t & (SEQ - 1);
  int bh = t >> 9;
  int h = bh & (HEADS - 1), b = bh >> 4;
  const float* up = u + h * DHEAD;
  const bf16* kp = Kb + ((size_t)(b * SEQ + j)) * DMODEL + h * DHEAD;
  float s = 0.f;
#pragma unroll 8
  for (int d = 0; d < DHEAD; ++d) s += up[d] * (float)kp[d];
  EFp[t] = s;
}

__global__ void gh_kernel(const float* __restrict__ v, const bf16* __restrict__ KR,
                          float* __restrict__ GHp) {
  int t = blockIdx.x * blockDim.x + threadIdx.x;
  if (t >= HEADS * SEQ) return;
  int j = t & (SEQ - 1);
  int h = t >> 9;
  const float* vp = v + h * DHEAD;
  const bf16* kp = KR + (size_t)j * DMODEL + h * DHEAD;
  float s = 0.f;
#pragma unroll 8
  for (int d = 0; d < DHEAD; ++d) s += vp[d] * (float)kp[d];
  GHp[t] = s;
}

// ---------------------------------------------------------------------------
// Exact Transformer-XL pad+reshape shift semantics for L == S:
//   j <= i   : raw[i, S-1+j-i]
//   j == i+1 : 0
//   j >  i+1 : raw[i+1, j-i-2]   (row clamped at the 32-row block edge)
// ---------------------------------------------------------------------------
__device__ __forceinline__ float bd_term(const bf16 bdr[32][SEQ + 16],
                                         int li, int i, int j) {
  if (j <= i) return (float)bdr[li][SEQ - 1 + j - i];
  if (j == i + 1) return 0.f;
  int lr = (li + 1 < 32) ? li + 1 : 31;
  return (float)bdr[lr][j - i - 2];
}

// ---------------------------------------------------------------------------
// Fused attention: one block = (b,h) x 32 query rows; 64 threads = 2 waves,
// each wave owns a 16-row q tile. Flash-style online softmax over j.
// ---------------------------------------------------------------------------
__global__ __launch_bounds__(64)
void attn_kernel(const bf16* __restrict__ Q, const bf16* __restrict__ Kb,
                 const bf16* __restrict__ Vb, const bf16* __restrict__ KR,
                 const float* __restrict__ EFp, const float* __restrict__ GHp,
                 bf16* __restrict__ Out) {
  __shared__ bf16 bdr[32][SEQ + 16];   // q·kr^T strip for this block's rows
  __shared__ bf16 vt[64][40];          // V tile transposed: vt[dh][j_loc]
  __shared__ bf16 pt[2][16][40];       // per-wave P staging (C-layout -> A-layout)

  const int lane = threadIdx.x & 31;
  const int wave = threadIdx.x >> 5;
  const int bh = blockIdx.y;
  const int b = bh >> 4, h = bh & (HEADS - 1);
  const int i0 = blockIdx.x * 32 + wave * 16;
  const float scale = 0.125f;          // 1/sqrt(64)

  const int Nl = lane & 15;
  const int Mb = (lane >> 4) << 3;

  // q fragments for this wave's 16 rows (K-chunks 0..31 and 32..63), reused
  // for both the BD strip and the AC scores.
  const bf16* qbase = Q + ((size_t)(b * SEQ + i0)) * DMODEL + h * DHEAD;
  bf16x16 qA0 = load_a_rm(qbase, DMODEL, lane);
  bf16x16 qA1 = load_a_rm(qbase + 32, DMODEL, lane);

  // ---- BD raw strip: bdr[loc_i][j] = q_i · kr_j  (32 j-tiles x 2 k-chunks) --
  for (int j0 = 0; j0 < SEQ; j0 += 16) {
    const bf16* krp = KR + (size_t)j0 * DMODEL + h * DHEAD;
    f32x8 c = {};
    c = wmma_bf16(qA0, load_b_nt(krp, DMODEL, lane), c);
    c = wmma_bf16(qA1, load_b_nt(krp + 32, DMODEL, lane), c);
#pragma unroll
    for (int r = 0; r < 8; ++r)
      bdr[wave * 16 + Mb + r][j0 + Nl] = (bf16)c[r];
  }
  __syncthreads();

  float mrun[8], lrun[8];
  f32x8 O[4] = {};
#pragma unroll
  for (int r = 0; r < 8; ++r) { mrun[r] = -3.0e38f; lrun[r] = 0.f; }

  const float* EFrow = EFp + (size_t)bh * SEQ;
  const float* GHrow = GHp + (size_t)h * SEQ;

  for (int j0 = 0; j0 < SEQ; j0 += 32) {
    __syncthreads();
    // Stage V[j0..j0+31][0..63] transposed into LDS (coalesced global reads).
    for (int idx = threadIdx.x; idx < 32 * 64; idx += 64) {
      int jl = idx >> 6, d = idx & 63;
      vt[d][jl] = Vb[((size_t)(b * SEQ + j0 + jl)) * DMODEL + h * DHEAD + d];
    }
    __syncthreads();

    // AC scores: two 16x16 j-tiles, each accumulated over DH=64 (2 wmma).
    const bf16* kb0 = Kb + ((size_t)(b * SEQ + j0)) * DMODEL + h * DHEAD;
    const bf16* kb1 = kb0 + (size_t)16 * DMODEL;
    f32x8 c0 = {}, c1 = {};
    c0 = wmma_bf16(qA0, load_b_nt(kb0, DMODEL, lane), c0);
    c0 = wmma_bf16(qA1, load_b_nt(kb0 + 32, DMODEL, lane), c0);
    c1 = wmma_bf16(qA0, load_b_nt(kb1, DMODEL, lane), c1);
    c1 = wmma_bf16(qA1, load_b_nt(kb1 + 32, DMODEL, lane), c1);

    float p0[8], p1[8];
#pragma unroll
    for (int r = 0; r < 8; ++r) {
      const int M = Mb + r;
      const int i = i0 + M;
      const int li = wave * 16 + M;
      const int ja = j0 + Nl, jb = j0 + 16 + Nl;
      float s0 = (c0[r] + bd_term(bdr, li, i, ja) + EFrow[ja] + GHrow[ja]) * scale;
      float s1 = (c1[r] + bd_term(bdr, li, i, jb) + EFrow[jb] + GHrow[jb]) * scale;

      // Row reduction across the 16 N-lanes (stays inside each 16-lane group).
      float mt = fmaxf(s0, s1);
#pragma unroll
      for (int mk = 1; mk < 16; mk <<= 1) mt = fmaxf(mt, __shfl_xor(mt, mk, 32));
      float mn = fmaxf(mrun[r], mt);
      float alpha = __expf(mrun[r] - mn);
      mrun[r] = mn;
      float e0 = __expf(s0 - mn), e1 = __expf(s1 - mn);
      float ps = e0 + e1;
#pragma unroll
      for (int mk = 1; mk < 16; mk <<= 1) ps += __shfl_xor(ps, mk, 32);
      lrun[r] = lrun[r] * alpha + ps;
#pragma unroll
      for (int n = 0; n < 4; ++n) O[n][r] *= alpha;
      p0[r] = e0; p1[r] = e1;
    }

    // P (C-layout f32) -> bf16 A-layout via per-wave LDS tile.
#pragma unroll
    for (int r = 0; r < 8; ++r) {
      pt[wave][Mb + r][Nl]      = (bf16)p0[r];
      pt[wave][Mb + r][16 + Nl] = (bf16)p1[r];
    }
    __builtin_amdgcn_wave_barrier();   // same-wave LDS ops are in-order

    bf16x16 pA = load_a_lds(&pt[wave][0][0], 40, lane);
#pragma unroll
    for (int n = 0; n < 4; ++n) {
      bf16x16 vB = load_b_nt(&vt[n * 16][0], 40, lane);
      O[n] = wmma_bf16(pA, vB, O[n]);
    }
  }

  // Finalize: divide by softmax denominator, store bf16 [b, i, h*64+dh].
#pragma unroll
  for (int r = 0; r < 8; ++r) {
    float inv = 1.0f / lrun[r];
    const int i = i0 + Mb + r;
#pragma unroll
    for (int n = 0; n < 4; ++n)
      Out[((size_t)(b * SEQ + i)) * DMODEL + h * DHEAD + n * 16 + Nl] =
          (bf16)(O[n][r] * inv);
  }
}

// ---------------------------------------------------------------------------
// Host-side orchestration (all launches on `stream`; ws-only scratch).
// Input order: x, Wq, Wk, Wv, Wo, Wkr, u, v, pos_emb
// ---------------------------------------------------------------------------
extern "C" void kernel_launch(void* const* d_in, const int* in_sizes, int n_in,
                              void* d_out, int out_size, void* d_ws, size_t ws_size,
                              hipStream_t stream) {
  (void)in_sizes; (void)n_in; (void)out_size; (void)ws_size;
  const float* x   = (const float*)d_in[0];
  const float* Wq  = (const float*)d_in[1];
  const float* Wk  = (const float*)d_in[2];
  const float* Wv  = (const float*)d_in[3];
  const float* Wo  = (const float*)d_in[4];
  const float* Wkr = (const float*)d_in[5];
  const float* u   = (const float*)d_in[6];
  const float* v   = (const float*)d_in[7];
  const float* pe  = (const float*)d_in[8];

  char* ws = (char*)d_ws;
  size_t off = 0;
  auto alloc = [&](size_t bytes) {
    void* p = ws + off;
    off += (bytes + 255) & ~(size_t)255;
    return p;
  };
  bf16* xb    = (bf16*)alloc((size_t)ROWS * DMODEL * 2);
  bf16* wqbT  = (bf16*)alloc((size_t)DMODEL * DMODEL * 2);  // [n][k] transposed
  bf16* wkbT  = (bf16*)alloc((size_t)DMODEL * DMODEL * 2);
  bf16* wvbT  = (bf16*)alloc((size_t)DMODEL * DMODEL * 2);
  bf16* wobT  = (bf16*)alloc((size_t)DMODEL * DMODEL * 2);
  bf16* wkrbT = (bf16*)alloc((size_t)DMODEL * DMODEL * 2);
  bf16* peb   = (bf16*)alloc((size_t)SEQ * DMODEL * 2);
  bf16* qbf   = (bf16*)alloc((size_t)ROWS * DMODEL * 2);
  bf16* kbf   = (bf16*)alloc((size_t)ROWS * DMODEL * 2);
  bf16* vbf   = (bf16*)alloc((size_t)ROWS * DMODEL * 2);
  bf16* krb   = (bf16*)alloc((size_t)SEQ * DMODEL * 2);
  bf16* attb  = (bf16*)alloc((size_t)ROWS * DMODEL * 2);
  float* efp  = (float*)alloc((size_t)BQ * HEADS * SEQ * 4);
  float* ghp  = (float*)alloc((size_t)HEADS * SEQ * 4);

  auto cvt = [&](const float* s, bf16* d, int n) {
    cvt_kernel<<<(n + 255) / 256, 256, 0, stream>>>(s, d, n);
  };
  auto cvtT = [&](const float* s, bf16* d) {  // 1024x1024 weight transpose
    cvt_t_kernel<<<(DMODEL * DMODEL + 255) / 256, 256, 0, stream>>>(s, d, DMODEL, DMODEL);
  };
  cvt(x,  xb,  ROWS * DMODEL);
  cvt(pe, peb, SEQ * DMODEL);
  cvtT(Wq,  wqbT);
  cvtT(Wk,  wkbT);
  cvtT(Wv,  wvbT);
  cvtT(Wo,  wobT);
  cvtT(Wkr, wkrbT);

  dim3 gp(DMODEL / 128, ROWS / 128);
  gemm_bf16<false><<<gp, 256, 0, stream>>>(xb, wqbT, qbf, ROWS, DMODEL, DMODEL);
  gemm_bf16<false><<<gp, 256, 0, stream>>>(xb, wkbT, kbf, ROWS, DMODEL, DMODEL);
  gemm_bf16<false><<<gp, 256, 0, stream>>>(xb, wvbT, vbf, ROWS, DMODEL, DMODEL);
  dim3 gk(DMODEL / 128, SEQ / 128);
  gemm_bf16<false><<<gk, 256, 0, stream>>>(peb, wkrbT, krb, SEQ, DMODEL, DMODEL);

  ef_kernel<<<(BQ * HEADS * SEQ + 255) / 256, 256, 0, stream>>>(u, kbf, efp);
  gh_kernel<<<(HEADS * SEQ + 255) / 256, 256, 0, stream>>>(v, krb, ghp);

  attn_kernel<<<dim3(SEQ / 32, BQ * HEADS), 64, 0, stream>>>(
      qbf, kbf, vbf, krb, efp, ghp, attb);

  gemm_bf16<true><<<gp, 256, 0, stream>>>(attb, wobT, d_out, ROWS, DMODEL, DMODEL);
}
